// CombinedFocalTverskyLoss_53171695125135
// MI455X (gfx1250) — matrix operand order repaired and checked by reading, not compiled, not run on hardware
//
#include <hip/hip_runtime.h>

typedef __attribute__((ext_vector_type(16))) _Float16 v16h;
typedef __attribute__((ext_vector_type(8)))  float    v8f;

#define NCLS   10
#define IMH    512
#define IMW    512
#define NBATCH 8
#define TW     32
#define TH     32
#define HALOW  34
#define HALOH  34
#define NHALO  (HALOW*HALOH)   /* 1156 */
#define NPIX   (TW*TH)         /* 1024 */

// ---------------------------------------------------------------------------
// Stage 1: async tile load -> in-place softmax -> focal -> pb/gb precompute
//          -> WMMA class-stat reductions.  One WG = one 32x32 tile (+1 halo).
// ---------------------------------------------------------------------------
__global__ __launch_bounds__(256)
void fused_loss_stage1(const float* __restrict__ logits,
                       const int*   __restrict__ targets,
                       float*       __restrict__ gAcc /* [2][16][16] */)
{
    __shared__ float          sProbs[NHALO][NCLS]; // raw logits, then probs (in-place)
    __shared__ float          sPb[NPIX][NCLS];     // 3x3 max-min per class
    __shared__ float          sFocal[NPIX];
    __shared__ float          sRed[2][8][32];
    __shared__ unsigned       sBbits[NPIX];        // b1bits | b2bits<<16
    __shared__ unsigned char  sInfo[NHALO];        // tc | valid<<4 ; 0xFF = OOB
    __shared__ float          sConst[2];           // {1.0f, 0.0f}

    const int tid  = threadIdx.x;
    const int lane = tid & 31;
    const int wv   = __builtin_amdgcn_readfirstlane(tid >> 5); // uniform wave id
    const int b    = blockIdx.z;
    const int x0   = blockIdx.x * TW - 1;
    const int y0   = blockIdx.y * TH - 1;

    for (int i = tid; i < 2*8*32; i += 256) ((float*)sRed)[i] = 0.0f;
    if (tid == 0) { sConst[0] = 1.0f; sConst[1] = 0.0f; }

    const long plane  = (long)IMH * IMW;
    const float* bIn  = logits  + (long)b * NCLS * plane;
    const int*   bTgt = targets + (long)b * plane;

    // -------- Phase 1a: async-copy halo logits into LDS; stage labels ------
    for (int hp = tid; hp < NHALO; hp += 256) {
        const int py = hp / HALOW, px = hp - py * HALOW;
        const int gy = y0 + py,    gx = x0 + px;
        const bool inImg = (gy >= 0) & (gy < IMH) & (gx >= 0) & (gx < IMW);
        if (inImg) {
            const long off = (long)gy * IMW + gx;
            const float* gsrc = bIn + off;
            #pragma unroll
            for (int c = 0; c < NCLS; ++c) {
                const unsigned ldsOff = (unsigned)(uintptr_t)(&sProbs[hp][c]);
                // CDNA5 async vector-memory -> LDS copy (ASYNCcnt tracked)
                asm volatile("global_load_async_to_lds_b32 %0, %1, off"
                             :: "v"(ldsOff), "v"(gsrc + (long)c * plane)
                             : "memory");
            }
            const int t     = bTgt[off];
            const int valid = (t != 255);
            const int tc    = valid ? t : (NCLS - 1);       // clip(255)->9
            sInfo[hp] = (unsigned char)(tc | (valid << 4));
        } else {
            #pragma unroll
            for (int c = 0; c < NCLS; ++c) sProbs[hp][c] = 0.0f;
            sInfo[hp] = 0xFF;
        }
    }
    asm volatile("s_wait_asynccnt 0x0" ::: "memory");
    __syncthreads();

    // -------- Phase 1b: in-place softmax + focal ---------------------------
    for (int hp = tid; hp < NHALO; hp += 256) {
        const int py = hp / HALOW, px = hp - py * HALOW;
        const unsigned char info = sInfo[hp];
        if (info != 0xFF) {
            float e[NCLS];
            float m = -3.4e38f;
            #pragma unroll
            for (int c = 0; c < NCLS; ++c) { e[c] = sProbs[hp][c]; m = fmaxf(m, e[c]); }
            float s = 0.0f;
            #pragma unroll
            for (int c = 0; c < NCLS; ++c) { e[c] = __expf(e[c] - m); s += e[c]; }
            const float inv = 1.0f / s;
            #pragma unroll
            for (int c = 0; c < NCLS; ++c) sProbs[hp][c] = e[c] * inv;

            if (py >= 1 && py <= TH && px >= 1 && px <= TW) { // interior
                const int ip = (py - 1) * TW + (px - 1);
                const int valid = (info >> 4) & 1;
                const int tc    = info & 0xF;
                float f = 0.0f;
                if (valid) {
                    const float pt = e[tc] * inv;
                    const float ce = -__logf(fmaxf(pt, 1e-30f));
                    f = 0.25f * (1.0f - pt) * ce;            // alpha=.25, gamma=1
                }
                sFocal[ip] = f;
            }
        }
    }
    __syncthreads();

    // -------- Phase 2: per-pixel B-operand bits + pb (uniform flow) --------
    for (int ip = tid; ip < NPIX; ip += 256) {
        const int py  = (ip >> 5) + 1, px = (ip & 31) + 1;
        const int hp0 = py * HALOW + px;
        int  nh[9];
        bool ok[9];
        unsigned lm = 0;
        #pragma unroll
        for (int t9 = 0; t9 < 9; ++t9) {
            const int dy = t9 / 3 - 1, dx = t9 % 3 - 1;
            nh[t9] = hp0 + dy * HALOW + dx;
            const unsigned char inf = sInfo[nh[t9]];
            ok[t9] = (inf != 0xFF);
            lm |= ok[t9] ? (1u << (inf & 0xFu)) : 0u;
        }
        const unsigned char cinf = sInfo[hp0];
        const unsigned v     = (cinf >> 4) & 1u;
        const unsigned tc    = cinf & 0xFu;
        const unsigned multi = ((lm & (lm - 1u)) != 0u) ? 1u : 0u;
        const unsigned b1bits = (1u << tc) | (1u << 10) | (v << 11);
        const unsigned b2bits = v ? ((multi ? lm : 0u) | (1u << 10)) : 0u;
        sBbits[ip] = b1bits | (b2bits << 16);

        #pragma unroll
        for (int c = 0; c < NCLS; ++c) {
            const float pc = sProbs[hp0][c];
            float mx = pc, mn = pc;
            #pragma unroll
            for (int t9 = 0; t9 < 9; ++t9) {
                if (t9 == 4) continue;
                const float pv = sProbs[nh[t9]][c];
                mx = ok[t9] ? fmaxf(mx, pv) : mx;
                mn = ok[t9] ? fminf(mn, pv) : mn;
            }
            sPb[ip][c] = mx - mn;                 // pb = max3 - min3 (cropped)
        }
    }
    __syncthreads();

    // -------- Phase 3: WMMA class-stat reductions --------------------------
    // Per-lane A-operand source descriptors (base ptr, per-chunk stride,
    // per-element stride) -> exactly one unconditional ds_load per element.
    v8f d1 = {}; v8f d2 = {};
    const int  rowA   = lane & 15;
    const int  colB   = lane & 15;
    const bool hiHalf = lane >= 16;
    const int  kBaseA = hiHalf ? 8 : 0;    // A: lanes0-15 K={0..7,16..23}
    const int  kBaseB = hiHalf ? 8 : 0;    // B: lanes0-15 K={0..7,16..23}

    const char* a1B; int a1Ch, a1E;
    const char* a2B; int a2Ch, a2E;
    if (rowA < NCLS) {
        a1B = (const char*)sProbs + (HALOW + 1) * (NCLS*4) + rowA*4;
        a1Ch = HALOW * (NCLS*4);  a1E = NCLS*4;          // probs row
        a2B = (const char*)sPb + rowA*4;
        a2Ch = TW * (NCLS*4);     a2E = NCLS*4;          // pb row
    } else if (rowA == 10) {
        a1B = (const char*)&sConst[0]; a1Ch = 0; a1E = 0; // 1.0
        a2B = (const char*)&sConst[0]; a2Ch = 0; a2E = 0; // 1.0
    } else if (rowA == 11) {
        a1B = (const char*)sFocal;     a1Ch = TW*4; a1E = 4; // focal
        a2B = (const char*)&sConst[1]; a2Ch = 0;    a2E = 0; // 0.0
    } else {
        a1B = (const char*)&sConst[1]; a1Ch = 0; a1E = 0;
        a2B = (const char*)&sConst[1]; a2Ch = 0; a2E = 0;
    }

    for (int ch = wv * 4; ch < wv * 4 + 4; ++ch) {       // uniform loop
        const char* a1Row = a1B + ch * a1Ch + kBaseA * a1E;
        const char* a2Row = a2B + ch * a2Ch + kBaseA * a2E;
        const unsigned* bRow = sBbits + ch * 32 + kBaseB;
        v16h a1, b1, a2, b2;
        #pragma unroll
        for (int j = 0; j < 16; ++j) {
            const int jc = (j < 8) ? j : j + 8;          // K offset within half
            const float a1v = *(const float*)(a1Row + jc * a1E);
            const float a2v = *(const float*)(a2Row + jc * a2E);
            const unsigned bits = bRow[jc];              // ds_load offset:jc*4
            const float b1v = ((bits >> colB) & 1u) ? 1.0f : 0.0f;
            const float b2v = ((bits >> (colB + 16)) & 1u) ? 1.0f : 0.0f;
            a1[j] = (_Float16)a1v;  a2[j] = (_Float16)a2v;
            b1[j] = (_Float16)b1v;  b2[j] = (_Float16)b2v;
        }
        d1 = __builtin_amdgcn_wmma_f32_16x16x32_f16(false, a1, false, b1,
                                                    (short)0, d1, false, false);
        d2 = __builtin_amdgcn_wmma_f32_16x16x32_f16(false, a2, false, b2,
                                                    (short)0, d2, false, false);
    }

    // cross-wave reduce D tiles in LDS
    #pragma unroll
    for (int r = 0; r < 8; ++r) {
        atomicAdd(&sRed[0][r][lane], d1[r]);
        atomicAdd(&sRed[1][r][lane], d2[r]);
    }
    __syncthreads();

    // global accumulate (D layout: VGPR r, lanes0-15 -> M=r, lanes16-31 -> M=r+8)
    for (int e = tid; e < 512; e += 256) {
        const int mat = e >> 8;
        const int r   = (e >> 5) & 7;
        const int ln  = e & 31;
        const int M   = r + ((ln >= 16) ? 8 : 0);
        const int N   = ln & 15;
        atomicAdd(&gAcc[mat * 256 + M * 16 + N], ((float*)sRed)[e]);
    }
}

// ---------------------------------------------------------------------------
// Stage 2: tiny finalize kernel -> scalar loss
// ---------------------------------------------------------------------------
__global__ void finalize_loss(const float* __restrict__ gAcc,
                              float* __restrict__ out)
{
    if (threadIdx.x == 0 && blockIdx.x == 0) {
        const float* D1 = gAcc;
        const float* D2 = gAcc + 256;
        const float focal_num = D1[11*16 + 11];
        const float vm_sum    = D1[10*16 + 11];
        const float focal_loss = focal_num / fmaxf(vm_sum, 1e-8f);
        float tsum = 0.0f, dsum = 0.0f;
        #pragma unroll
        for (int c = 0; c < NCLS; ++c) {
            const float TP = D1[c*16 + c];
            const float S  = D1[c*16 + 10];
            const float Nc = D1[10*16 + c];
            const float FP = S - TP, FN = Nc - TP;
            tsum += (TP + 1e-6f) / (TP + 0.5f*FP + 0.7f*FN + 1e-6f);
            const float inter = D2[c*16 + c];
            const float dP    = D2[c*16 + 10];
            const float dG    = D2[10*16 + c];
            dsum += (2.0f*inter + 1e-6f) / (dP + dG + 1e-6f);
        }
        const float tversky_loss  = 1.0f - 0.1f * tsum;
        const float boundary_dice = 1.0f - 0.1f * dsum;
        out[0] = 0.5f*focal_loss + 0.5f*tversky_loss + 0.2f*boundary_dice;
    }
}

// ---------------------------------------------------------------------------
extern "C" void kernel_launch(void* const* d_in, const int* in_sizes, int n_in,
                              void* d_out, int out_size, void* d_ws, size_t ws_size,
                              hipStream_t stream)
{
    (void)in_sizes; (void)n_in; (void)out_size; (void)ws_size;
    const float* logits  = (const float*)d_in[0];
    const int*   targets = (const int*)d_in[1];
    float*       gAcc    = (float*)d_ws;

    hipMemsetAsync(d_ws, 0, 2 * 256 * sizeof(float), stream);

    dim3 grid(IMW / TW, IMH / TH, NBATCH);   // 16 x 16 x 8 = 2048 tiles
    fused_loss_stage1<<<grid, 256, 0, stream>>>(logits, targets, gAcc);
    finalize_loss<<<1, 32, 0, stream>>>(gAcc, (float*)d_out);
}
